// MultiHeadGraphAttention_1211180778440
// MI455X (gfx1250) — compile-verified, matrix operand order checked by reading, not run on hardware
//
#include <hip/hip_runtime.h>

// MI455X / gfx1250, wave32. One workgroup per (b,t) slice: B*T = 1024 blocks, 256 threads (8 waves).

#define NN   128   // nodes
#define DD   64    // feature dim (D == E)
#define NHEAD 8
#define HD   8     // head dim
#define XP   72    // pitch (bf16 elems) for 64-wide row-major LDS matrices
#define VTP  136   // pitch (bf16 elems) for transposed V over m
#define SCP  132   // pitch (f32 elems) for scores

typedef __attribute__((ext_vector_type(16))) __bf16 v16bf;
typedef __attribute__((ext_vector_type(2)))  __bf16 v2bf;
typedef __attribute__((ext_vector_type(8)))  float  v8f;
typedef unsigned int   v4u  __attribute__((ext_vector_type(4), may_alias));
typedef unsigned int   v2u  __attribute__((ext_vector_type(2), may_alias));
typedef float          v4fa __attribute__((ext_vector_type(4), may_alias));
typedef int            v4ia __attribute__((ext_vector_type(4), may_alias));
typedef unsigned short aus  __attribute__((may_alias));

union ABu { v4u u[2]; v16bf v; };

static __device__ __forceinline__ unsigned short f2bf(float f) {
  return __builtin_bit_cast(unsigned short, (__bf16)f);   // native v_cvt on gfx1250
}

// Two floats -> one packed bf16x2 dword (clang fuses into a single v_cvt_pk_bf16_f32).
static __device__ __forceinline__ unsigned pk2(float a, float b) {
  v2bf v = {(__bf16)a, (__bf16)b};
  return __builtin_bit_cast(unsigned, v);
}

static __device__ __forceinline__ v8f splat8(float x) {
  v8f v = {x, x, x, x, x, x, x, x};
  return v;
}

// Load a 16x32 bf16 WMMA A/B fragment from a row-major LDS matrix.
// ISA layout: lanes 0-15 row=r0+li, K=0..7 & 16..23; lanes 16-31 row=r0+li, K=8..15 & 24..31.
static __device__ __forceinline__ v16bf ld16x32(const unsigned short* m, int pitch,
                                                int r0, int c0, int li, int half) {
  const aus* row = (const aus*)(m + (r0 + li) * pitch + c0 + half * 8);
  ABu t;
  t.u[0] = *(const v4u*)row;
  t.u[1] = *(const v4u*)(row + 16);
  return t.v;
}

// Same, but source matrix logically has only K=0..7 valid (hd=8), K=8..31 zero.
static __device__ __forceinline__ v16bf ld16x32_k8(const unsigned short* m, int pitch,
                                                   int r0, int c0, int li, int half) {
  ABu t;
  v4u z = {0u, 0u, 0u, 0u};
  t.u[0] = z; t.u[1] = z;
  if (half == 0) t.u[0] = *(const v4u*)(m + (r0 + li) * pitch + c0);
  return t.v;
}

#define WMMA_BF16(a, b, c) \
  __builtin_amdgcn_wmma_f32_16x16x32_bf16(false, (a), false, (b), (short)0, (c), false, false)

// ---- LDS layout (bytes), phase-overlapped. Total 152576 B (< 320 KB/WGP). ----
// [0,67584)       : scores f32 [128][132]  (phase 2); overlays:
//   [0,18432)     :   xbf  [128][72] bf16  (phases 0-1)
//   [18432,27648) :   Wq^T [64][72] bf16   (phases 0-1)
//   [27648,36864) :   Wk^T
//   [36864,46080) :   Wv^T
// [67584,76800)   : Wo^T [64][72] bf16
// [76800,77824)   : biases f32 (bq,bk,bv,bo) x 64
// [77824,96256)   : Q bf16 [128][72]  (pre-scaled by 1/sqrt(hd))
// [96256,114688)  : K bf16 [128][72]
// [114688,132096) : V^T bf16 [64][136]
// [132096,134144) : adj bitmask u32 [128][4]
// [134144,152576) : O bf16 [128][72]  (attn output, pre final projection)

__global__ __launch_bounds__(256) void
mhga_kernel(const float* __restrict__ x, const int* __restrict__ adj,
            const float* __restrict__ Wq, const float* __restrict__ bq,
            const float* __restrict__ Wk, const float* __restrict__ bk,
            const float* __restrict__ Wv, const float* __restrict__ bv,
            const float* __restrict__ Wo, const float* __restrict__ bo,
            float* __restrict__ out) {
  __shared__ __align__(16) unsigned char smem[152576];

  float*          sc   = (float*)(smem);
  unsigned short* xbf  = (unsigned short*)(smem);
  unsigned short* wqt  = (unsigned short*)(smem + 18432);
  unsigned short* wkt  = (unsigned short*)(smem + 27648);
  unsigned short* wvt  = (unsigned short*)(smem + 36864);
  unsigned short* wot  = (unsigned short*)(smem + 67584);
  float*          bias = (float*)(smem + 76800);
  unsigned short* qbf  = (unsigned short*)(smem + 77824);
  unsigned short* kbf  = (unsigned short*)(smem + 96256);
  unsigned short* vt   = (unsigned short*)(smem + 114688);
  unsigned*       adjb = (unsigned*)(smem + 132096);
  unsigned short* obf  = (unsigned short*)(smem + 134144);

  const int tid  = threadIdx.x;
  const int lane = tid & 31;
  const int wid  = tid >> 5;
  const int li   = lane & 15;
  const int half = lane >> 4;

  // ---------------- Phase 0: stage inputs into LDS (b128 global loads) ----------------
  const float* xg = x + (size_t)blockIdx.x * (NN * DD);
  const v4fa* xg4 = (const v4fa*)xg;
#pragma unroll
  for (int i = 0; i < 8; ++i) {                // 2048 float4s of x
    int q = tid + 256 * i;
    int r = q >> 4, c4 = (q & 15) * 4;
    v4fa v = xg4[q];
    v2u w = {pk2(v.x, v.y), pk2(v.z, v.w)};    // 2x v_cvt_pk_bf16_f32
    *(v2u*)(xbf + r * XP + c4) = w;
  }
  const float qs = 0.35355339059327379f;       // 1/sqrt(hd): fold into Wq, bq
#pragma unroll
  for (int i = 0; i < 4; ++i) {                // 1024 float4s per weight matrix
    int q = tid + 256 * i;
    int d = q >> 4, e4 = (q & 15) * 4;
    v4fa a;
    a = ((const v4fa*)Wq)[q];
    wqt[(e4 + 0) * XP + d] = f2bf(a.x * qs); wqt[(e4 + 1) * XP + d] = f2bf(a.y * qs);
    wqt[(e4 + 2) * XP + d] = f2bf(a.z * qs); wqt[(e4 + 3) * XP + d] = f2bf(a.w * qs);
    a = ((const v4fa*)Wk)[q];
    wkt[(e4 + 0) * XP + d] = f2bf(a.x); wkt[(e4 + 1) * XP + d] = f2bf(a.y);
    wkt[(e4 + 2) * XP + d] = f2bf(a.z); wkt[(e4 + 3) * XP + d] = f2bf(a.w);
    a = ((const v4fa*)Wv)[q];
    wvt[(e4 + 0) * XP + d] = f2bf(a.x); wvt[(e4 + 1) * XP + d] = f2bf(a.y);
    wvt[(e4 + 2) * XP + d] = f2bf(a.z); wvt[(e4 + 3) * XP + d] = f2bf(a.w);
    a = ((const v4fa*)Wo)[q];
    wot[(e4 + 0) * XP + d] = f2bf(a.x); wot[(e4 + 1) * XP + d] = f2bf(a.y);
    wot[(e4 + 2) * XP + d] = f2bf(a.z); wot[(e4 + 3) * XP + d] = f2bf(a.w);
  }
  if (tid < 64) {
    bias[tid]       = bq[tid] * qs;
    bias[64 + tid]  = bk[tid];
    bias[128 + tid] = bv[tid];
    bias[192 + tid] = bo[tid];
  }
#pragma unroll
  for (int i = 0; i < 2; ++i) {                // adjacency bitmask: 128 rows x 4 words
    int w = tid + 256 * i;
    int n = w >> 2, g = w & 3;
    const v4ia* ap = (const v4ia*)(adj + n * NN + g * 32);
    unsigned bits = 0u;
#pragma unroll
    for (int c = 0; c < 8; ++c) {
      v4ia a = ap[c];
      bits |= (a.x > 0 ? 1u : 0u) << (4 * c + 0);
      bits |= (a.y > 0 ? 1u : 0u) << (4 * c + 1);
      bits |= (a.z > 0 ? 1u : 0u) << (4 * c + 2);
      bits |= (a.w > 0 ? 1u : 0u) << (4 * c + 3);
    }
    adjb[w] = bits;
  }
  __syncthreads();

  // ---------------- Phase 1: Q/K/V projections (WMMA bf16) ----------------
  // 96 output tiles (3 mats x 8 Mtiles x 4 Ntiles), strided over 8 waves.
#pragma unroll
  for (int i = 0; i < 12; ++i) {
    int t = wid + 8 * i;
    int mat = t >> 5, r = t & 31, mt = r >> 2, nt = r & 3;
    const unsigned short* wmat = (mat == 0) ? wqt : (mat == 1) ? wkt : wvt;
    v8f c = splat8(bias[mat * 64 + nt * 16 + li]);
#pragma unroll
    for (int kb = 0; kb < DD; kb += 32) {
      v16bf a = ld16x32(xbf,  XP, mt * 16, kb, li, half);
      v16bf b = ld16x32(wmat, XP, nt * 16, kb, li, half);
      c = WMMA_BF16(a, b, c);
    }
    if (mat < 2) {                              // Q, K: row-major bf16 scatter
      unsigned short* dst = (mat == 0) ? qbf : kbf;
#pragma unroll
      for (int j = 0; j < 8; ++j)
        dst[(mt * 16 + j + 8 * half) * XP + nt * 16 + li] = f2bf(c[j]);
    } else {                                    // V: transposed -> one b128 store/lane
      v4u pk = {pk2(c[0], c[1]), pk2(c[2], c[3]),
                pk2(c[4], c[5]), pk2(c[6], c[7])};   // 4x v_cvt_pk_bf16_f32
      *(v4u*)(vt + (nt * 16 + li) * VTP + mt * 16 + 8 * half) = pk;
    }
  }
  __syncthreads();

  // ---------------- Phase 2: per-head attention ----------------
  for (int h = 0; h < NHEAD; ++h) {
    // (a) scores = Qh @ Kh^T (hd=8 zero-padded into K=32), mask, f32 -> LDS
#pragma unroll
    for (int i = 0; i < 8; ++i) {
      int mt = i, mct = wid;                    // row-tile i, col-tile = wave id
      v16bf a = ld16x32_k8(qbf, XP, mt * 16, h * HD, li, half);
      v16bf b = ld16x32_k8(kbf, XP, mct * 16, h * HD, li, half);
      v8f c = splat8(0.0f);
      c = WMMA_BF16(a, b, c);
#pragma unroll
      for (int j = 0; j < 8; ++j) {
        int n = mt * 16 + j + 8 * half;
        int m = mct * 16 + li;
        unsigned bit = (adjb[n * 4 + (m >> 5)] >> (m & 31)) & 1u;
        sc[n * SCP + m] = bit ? c[j] : -3.0e38f;
      }
    }
    __syncthreads();

    // (b) softmax over m, one thread per row (vectorized x4); bf16 attn written in place
    if (tid < NN) {
      float* row = sc + tid * SCP;
      v4fa* r4 = (v4fa*)row;
      float mx = -3.0e38f;
      for (int m = 0; m < 32; ++m) {
        v4fa r = r4[m];
        mx = fmaxf(mx, fmaxf(fmaxf(r.x, r.y), fmaxf(r.z, r.w)));
      }
      float sum = 0.0f;
      for (int m = 0; m < 32; ++m) {
        v4fa r = r4[m];
        r.x = __expf(r.x - mx); r.y = __expf(r.y - mx);
        r.z = __expf(r.z - mx); r.w = __expf(r.w - mx);
        sum += (r.x + r.y) + (r.z + r.w);
        r4[m] = r;
      }
      float inv = 1.0f / sum;
      for (int m = 0; m < 32; ++m) {
        v4fa r = r4[m];
        v2u w = {pk2(r.x * inv, r.y * inv), pk2(r.z * inv, r.w * inv)};
        *(v2u*)((aus*)row + 4 * m) = w;        // in-place: write never passes read
      }
    }
    __syncthreads();

    // (c) out_h = attn @ Vh : 8 M-tiles, one per wave, K=128
    {
      int mt = wid;
      const aus* att = (const aus*)sc;
      const int apitch = SCP * 2;               // bf16 pitch of in-place attn rows
      v8f c = splat8(0.0f);
#pragma unroll
      for (int kb = 0; kb < NN; kb += 32) {
        ABu ta, tb;
        const aus* arow = att + (mt * 16 + li) * apitch + kb + half * 8;
        ta.u[0] = *(const v4u*)arow;
        ta.u[1] = *(const v4u*)(arow + 16);
        int vr = h * HD + li; if (vr > DD - 1) vr = DD - 1;   // cols >= 8 discarded below
        const aus* brow = (const aus*)(vt + vr * VTP + kb + half * 8);
        tb.u[0] = *(const v4u*)brow;
        tb.u[1] = *(const v4u*)(brow + 16);
        c = WMMA_BF16(ta.v, tb.v, c);
      }
      if (li < HD) {
#pragma unroll
        for (int j = 0; j < 8; ++j)
          obf[(mt * 16 + j + 8 * half) * XP + h * HD + li] = f2bf(c[j]);
      }
    }
    __syncthreads();
  }

  // ---------------- Phase 3: output projection (WMMA bf16), fp32 store ----------------
  float* og = out + (size_t)blockIdx.x * (NN * DD);
#pragma unroll
  for (int i = 0; i < 4; ++i) {
    int t = wid + 8 * i;
    int mt = t >> 2, nt = t & 3;
    v8f c = splat8(bias[3 * 64 + nt * 16 + li]);
#pragma unroll
    for (int kb = 0; kb < DD; kb += 32) {
      v16bf a = ld16x32(obf, XP, mt * 16, kb, li, half);
      v16bf b = ld16x32(wot, XP, nt * 16, kb, li, half);
      c = WMMA_BF16(a, b, c);
    }
#pragma unroll
    for (int j = 0; j < 8; ++j)
      og[(mt * 16 + j + 8 * half) * DD + nt * 16 + li] = c[j];
  }
}

extern "C" void kernel_launch(void* const* d_in, const int* in_sizes, int n_in,
                              void* d_out, int out_size, void* d_ws, size_t ws_size,
                              hipStream_t stream) {
  (void)n_in; (void)d_ws; (void)ws_size; (void)out_size;
  const float* x   = (const float*)d_in[0];
  const int*   adj = (const int*)d_in[1];
  const float* Wq  = (const float*)d_in[2];
  const float* bq  = (const float*)d_in[3];
  const float* Wk  = (const float*)d_in[4];
  const float* bk  = (const float*)d_in[5];
  const float* Wv  = (const float*)d_in[6];
  const float* bv  = (const float*)d_in[7];
  const float* Wo  = (const float*)d_in[8];
  const float* bo  = (const float*)d_in[9];
  float* out = (float*)d_out;

  int blocks = in_sizes[0] / (NN * DD);        // B*T = 1024
  mhga_kernel<<<blocks, 256, 0, stream>>>(x, adj, Wq, bq, Wk, bk, Wv, bv, Wo, bo, out);
}
